// Conv_DCFD_55001351192652
// MI455X (gfx1250) — matrix operand order repaired and checked by reading, not compiled, not run on hardware
//
#include <hip/hip_runtime.h>
#include <hip/hip_bf16.h>
#include <math.h>

// ---------------------------------------------------------------------------
// Types for gfx1250 WMMA (wave32): D = A(16x32 bf16) x B(32x16 bf16) + C(f32)
// ---------------------------------------------------------------------------
typedef __bf16 bf16;
typedef __attribute__((ext_vector_type(16))) __bf16 v16bf;
typedef __attribute__((ext_vector_type(8)))  __bf16 v8bf;
typedef __attribute__((ext_vector_type(2)))  __bf16 v2bf;
typedef __attribute__((ext_vector_type(8)))  float  v8f;

#define NB    16
#define CINX  128
#define INTER 64
#define BSZ   36
#define OCH   128
#define HH    64
#define WW    64
#define HWPX  (HH * WW)

#define STRB  784   // boBT row stride (768 + 16 pad), keeps 16B align, skews banks
#define STRC  40    // conv BchT row stride (32 + 8 pad)

// ---------------------------------------------------------------------------
// Fragment loaders per CDNA5 ISA 7.12.2 VGPR layouts (wave32).
// A (16-bit, 16x32 MxK): lane L: M=L&15; slots 0..7 hold K=lh*8+0..7,
//   slots 8..15 hold K=16+lh*8+0..7  ->  two contiguous 16B loads.
// B (16-bit, 32x16 KxN): lane L: N=L&15; slots 0..7 hold K=lh*8+0..7,
//   slots 8..15 hold K=16+lh*8+0..7. With the tile stored TRANSPOSED [n][k],
//   that's two contiguous 16B loads.
// C/D (32-bit, 16x16): lane L, VGPR r: M = r + 8*(L>>4), N = L&15.
// ---------------------------------------------------------------------------
__device__ __forceinline__ v16bf load_a_frag(const bf16* __restrict__ A, int lda) {
  int lane = threadIdx.x & 31;
  int m  = lane & 15;
  int lh = lane >> 4;
  const bf16* row = A + (size_t)m * lda + lh * 8;
  v8bf a0 = *(const v8bf*)(row);
  v8bf a1 = *(const v8bf*)(row + 16);
  return __builtin_shufflevector(a0, a1, 0, 1, 2, 3, 4, 5, 6, 7,
                                         8, 9, 10, 11, 12, 13, 14, 15);
}

// Bt points at [n=0][k_chunk_base] of a row-major transposed tile [N][K].
__device__ __forceinline__ v16bf load_b_frag_t(const bf16* Bt, int ldk) {
  int lane = threadIdx.x & 31;
  int n  = lane & 15;
  int lh = lane >> 4;
  const bf16* row = Bt + (size_t)n * ldk + lh * 8;
  v8bf b0 = *(const v8bf*)(row);
  v8bf b1 = *(const v8bf*)(row + 16);
  return __builtin_shufflevector(b0, b1, 0, 1, 2, 3, 4, 5, 6, 7,
                                         8, 9, 10, 11, 12, 13, 14, 15);
}

// ---------------------------------------------------------------------------
// Prep: fold BN, reorder conv weights to [oc][l*C + c] (so K-chunks of 32
// align to channel chunks), convert weights/coef to bf16.
// ---------------------------------------------------------------------------
__global__ void k_prep(const float* __restrict__ w1, const float* __restrict__ w2,
                       const float* __restrict__ coefw,
                       const float* __restrict__ g1, const float* __restrict__ be1,
                       const float* __restrict__ m1, const float* __restrict__ v1,
                       const float* __restrict__ g2, const float* __restrict__ be2,
                       const float* __restrict__ m2, const float* __restrict__ v2,
                       bf16* __restrict__ w1r, bf16* __restrict__ w2r,
                       bf16* __restrict__ coefbf,
                       float* __restrict__ s1t1, float* __restrict__ s2t2) {
  int tid = blockIdx.x * blockDim.x + threadIdx.x;
  int stride = gridDim.x * blockDim.x;
  for (int i = tid; i < INTER * 9 * CINX; i += stride) {
    int oc = i / (9 * CINX);
    int rem = i % (9 * CINX);
    int l = rem / CINX, c = rem % CINX;
    w1r[i] = (bf16)w1[(oc * CINX + c) * 9 + l];
  }
  for (int i = tid; i < 48 * 9 * INTER; i += stride) {
    int oc = i / (9 * INTER);
    int rem = i % (9 * INTER);
    int l = rem / INTER, c = rem % INTER;
    w2r[i] = (oc < BSZ) ? (bf16)w2[(oc * INTER + c) * 9 + l] : (bf16)0.0f;
  }
  for (int i = tid; i < OCH * (CINX * 6); i += stride) coefbf[i] = (bf16)coefw[i];
  for (int i = tid; i < INTER; i += stride) {
    float s = g1[i] * rsqrtf(v1[i] + 1e-3f);
    s1t1[i] = s;
    s1t1[INTER + i] = be1[i] - m1[i] * s;
  }
  for (int i = tid; i < BSZ; i += stride) {
    float s = g2[i] * rsqrtf(v2[i] + 1e-3f);
    s2t2[i] = s;
    s2t2[BSZ + i] = be2[i] - m2[i] * s;
  }
}

__global__ void k_cvt_x(const float* __restrict__ x, bf16* __restrict__ xbf, int n) {
  int i = blockIdx.x * blockDim.x + threadIdx.x;
  if (i < n) xbf[i] = (bf16)x[i];
}

// ---------------------------------------------------------------------------
// 3x3 SAME conv + BN + tanh as implicit GEMM with bf16 WMMA.
// Block = (n, row, 32-pixel tile). 8 waves; wave w -> (mt=w>>1, nt=w&1)
// output tile 16 oc x 16 px. K = 9*CI, chunked 32 (one tap l, 32 channels).
// Im2col chunk staged TRANSPOSED in LDS: BchT[pixel][k].
// ---------------------------------------------------------------------------
template <int CI, int COP, int COR, bool OBF>
__global__ __launch_bounds__(256) void k_conv3x3(const bf16* __restrict__ src,
                                                 const bf16* __restrict__ wr,
                                                 const float* __restrict__ st,
                                                 void* __restrict__ dst) {
  __shared__ bf16 BchT[32 * STRC];
  int blk  = blockIdx.x;
  int wt   = blk & 1;
  int hrow = (blk >> 1) & 63;
  int n    = blk >> 7;
  int w0   = wt * 32;

  int p    = threadIdx.x & 31;
  int krow = threadIdx.x >> 5;
  int wave = threadIdx.x >> 5;
  int lane = threadIdx.x & 31;
  constexpr int TILES = (COP / 16) * 2;
  int mt = wave >> 1, nt = wave & 1;

  v8f acc = {};

#pragma unroll 1
  for (int l = 0; l < 9; ++l) {
    int dh = l / 3 - 1, dw = l % 3 - 1;
    int row = hrow + dh;
#pragma unroll 1
    for (int cc = 0; cc < CI / 32; ++cc) {
      // cooperative im2col chunk: 32 channels x 32 pixels (coalesced reads)
#pragma unroll
      for (int e = 0; e < 4; ++e) {
        int kl  = krow + e * 8;
        int c   = cc * 32 + kl;
        int col = w0 + p + dw;
        bf16 val = (bf16)0.0f;
        if ((unsigned)row < (unsigned)HH && (unsigned)col < (unsigned)WW)
          val = src[((size_t)(n * CI + c) * HWPX) + row * WW + col];
        BchT[p * STRC + kl] = val;
      }
      __syncthreads();
      if (wave < TILES) {
        v16bf a = load_a_frag(wr + (size_t)(mt * 16) * (9 * CI) + (l * CI + cc * 32),
                              9 * CI);
        v16bf b = load_b_frag_t(BchT + nt * 16 * STRC, STRC);
        acc = __builtin_amdgcn_wmma_f32_16x16x32_bf16(false, a, false, b,
                                                      (short)0, acc, false, false);
      }
      __syncthreads();
    }
  }

  if (wave < TILES) {
    int nloc = lane & 15, lh = lane >> 4;
    int px = w0 + nt * 16 + nloc;
#pragma unroll
    for (int r = 0; r < 8; ++r) {
      int oc = mt * 16 + lh * 8 + r;
      if (oc < COR) {
        float val = acc[r] * st[oc] + st[COR + oc];
        val = tanhf(val);
        size_t idx = ((size_t)(n * COR + oc) * HWPX) + hrow * WW + px;
        if (OBF) ((bf16*)dst)[idx] = (bf16)val;
        else     ((float*)dst)[idx] = val;
      }
    }
  }
}

// ---------------------------------------------------------------------------
// Fused: bases (VALU, tiny) -> bo (VALU, 0.9 GF) -> out = coef @ bo (WMMA,
// 12.9 GF) + bias. bo never leaves LDS (avoids a 201 MB fp32 intermediate).
// bo staged TRANSPOSED: boBT[pixel][k = c*6+m].
// ---------------------------------------------------------------------------
__global__ __launch_bounds__(256) void k_fused(const float* __restrict__ x,
                                               const float* __restrict__ b36,
                                               const float* __restrict__ fb,
                                               const bf16* __restrict__ coefbf,
                                               const float* __restrict__ bias,
                                               float* __restrict__ out) {
  __shared__ float basesL[54 * 32];      // [l*6+m][pixel]
  __shared__ bf16  boBT[32 * STRB];      // transposed B: [pixel][k]

  int blk  = blockIdx.x;
  int wt   = blk & 1;
  int hrow = (blk >> 1) & 63;
  int n    = blk >> 7;
  int w0   = wt * 32;
  int t    = threadIdx.x;

  // ---- phase 1: bases[m][l][p] = sum_tem bcoef[m][tem][p] * fb[tem][l] ----
  if (t < 192) {
    int p = t & 31, m = t >> 5;
    float bc[6];
#pragma unroll
    for (int k = 0; k < 6; ++k)
      bc[k] = b36[((size_t)(n * BSZ + m * 6 + k) * HWPX) + hrow * WW + w0 + p];
#pragma unroll
    for (int l = 0; l < 9; ++l) {
      float s = 0.f;
#pragma unroll
      for (int k = 0; k < 6; ++k) s += bc[k] * fb[k * 9 + l];
      basesL[(l * 6 + m) * 32 + p] = s;
    }
  }
  __syncthreads();

  // ---- phase 2: bo[c][m][p] = sum_l bases[m][l][p] * x[c][p+tap_l] --------
  {
    int p = t & 31, g = t >> 5;
    float bl[54];
#pragma unroll
    for (int i = 0; i < 54; ++i) bl[i] = basesL[i * 32 + p];
#pragma unroll 1
    for (int ci = 0; ci < 16; ++ci) {
      int c = g * 16 + ci;
      const float* xc = x + (size_t)(n * CINX + c) * HWPX;
      float tap[9];
#pragma unroll
      for (int l = 0; l < 9; ++l) {
        int row = hrow + l / 3 - 1;
        int col = w0 + p + (l % 3) - 1;
        tap[l] = ((unsigned)row < (unsigned)HH && (unsigned)col < (unsigned)WW)
                     ? xc[row * WW + col] : 0.f;
      }
      float acc[6] = {0.f, 0.f, 0.f, 0.f, 0.f, 0.f};
#pragma unroll
      for (int l = 0; l < 9; ++l)
#pragma unroll
        for (int m = 0; m < 6; ++m) acc[m] += tap[l] * bl[l * 6 + m];
      // packed paired-bf16 stores (4B each, 4B aligned: 12c, stride 1568B)
#pragma unroll
      for (int j = 0; j < 3; ++j) {
        v2bf pk;
        pk[0] = (bf16)acc[2 * j];
        pk[1] = (bf16)acc[2 * j + 1];
        *(v2bf*)(&boBT[(size_t)p * STRB + c * 6 + 2 * j]) = pk;
      }
    }
  }
  __syncthreads();

  // ---- phase 3: out[128 x 32px] = coef[128 x 768] @ boBT^T[768 x 32] ------
  int wave = t >> 5;
  int lane = t & 31;
  int mt = wave;                      // 8 waves -> 8 M-tiles of 16 out-chans
  v8f acc0 = {};
  v8f acc1 = {};
  const bf16* A = coefbf + (size_t)(mt * 16) * 768;
#pragma unroll 2
  for (int kk = 0; kk < 24; ++kk) {
    v16bf a  = load_a_frag(A + kk * 32, 768);
    v16bf b0 = load_b_frag_t(boBT + kk * 32, STRB);
    v16bf b1 = load_b_frag_t(boBT + 16 * STRB + kk * 32, STRB);
    acc0 = __builtin_amdgcn_wmma_f32_16x16x32_bf16(false, a, false, b0,
                                                   (short)0, acc0, false, false);
    acc1 = __builtin_amdgcn_wmma_f32_16x16x32_bf16(false, a, false, b1,
                                                   (short)0, acc1, false, false);
  }
  int nloc = lane & 15, lh = lane >> 4;
#pragma unroll
  for (int r = 0; r < 8; ++r) {
    int o = mt * 16 + lh * 8 + r;
    float bo = bias[o];
    size_t base = ((size_t)(n * OCH + o) * HWPX) + hrow * WW + w0;
    out[base + nloc]      = acc0[r] + bo;
    out[base + 16 + nloc] = acc1[r] + bo;
  }
}

// ---------------------------------------------------------------------------
extern "C" void kernel_launch(void* const* d_in, const int* in_sizes, int n_in,
                              void* d_out, int out_size, void* d_ws, size_t ws_size,
                              hipStream_t stream) {
  (void)in_sizes; (void)n_in; (void)out_size; (void)ws_size;
  const float* x     = (const float*)d_in[0];
  const float* w1    = (const float*)d_in[1];
  const float* g1    = (const float*)d_in[2];
  const float* be1   = (const float*)d_in[3];
  const float* m1    = (const float*)d_in[4];
  const float* v1    = (const float*)d_in[5];
  const float* w2    = (const float*)d_in[6];
  const float* g2    = (const float*)d_in[7];
  const float* be2   = (const float*)d_in[8];
  const float* m2    = (const float*)d_in[9];
  const float* v2    = (const float*)d_in[10];
  const float* fb    = (const float*)d_in[11];
  const float* coefw = (const float*)d_in[12];
  const float* bias  = (const float*)d_in[13];
  float* out = (float*)d_out;

  char* ws = (char*)d_ws;
  size_t off = 0;
  auto alloc = [&](size_t bytes) -> void* {
    void* p = (void*)(ws + off);
    off += (bytes + 255) & ~(size_t)255;
    return p;
  };
  bf16*  xbf    = (bf16*)alloc((size_t)NB * CINX * HWPX * sizeof(bf16));   // 16.8 MB
  bf16*  hbf    = (bf16*)alloc((size_t)NB * INTER * HWPX * sizeof(bf16));  //  8.4 MB
  float* b36    = (float*)alloc((size_t)NB * BSZ * HWPX * sizeof(float));  //  9.4 MB
  bf16*  w1r    = (bf16*)alloc((size_t)INTER * 9 * CINX * sizeof(bf16));
  bf16*  w2r    = (bf16*)alloc((size_t)48 * 9 * INTER * sizeof(bf16));
  bf16*  coefbf = (bf16*)alloc((size_t)OCH * CINX * 6 * sizeof(bf16));
  float* s1t1   = (float*)alloc(2 * INTER * sizeof(float));
  float* s2t2   = (float*)alloc(2 * BSZ * sizeof(float));

  k_prep<<<128, 256, 0, stream>>>(w1, w2, coefw, g1, be1, m1, v1, g2, be2, m2, v2,
                                  w1r, w2r, coefbf, s1t1, s2t2);

  int nx = NB * CINX * HWPX;
  k_cvt_x<<<(nx + 255) / 256, 256, 0, stream>>>(x, xbf, nx);

  int nblk = NB * HH * (WW / 32);  // 2048
  k_conv3x3<CINX, 64, 64, true><<<nblk, 256, 0, stream>>>(xbf, w1r, s1t1, (void*)hbf);
  k_conv3x3<INTER, 48, BSZ, false><<<nblk, 256, 0, stream>>>(hbf, w2r, s2t2, (void*)b36);
  k_fused<<<nblk, 256, 0, stream>>>(x, b36, fb, coefbf, bias, out);
}
